// QBNNTransformerBlockLayered_25718264169096
// MI455X (gfx1250) — matrix operand order repaired and checked by reading, not compiled, not run on hardware
//
#include <hip/hip_runtime.h>
#include <math.h>

// QBNN transformer block for gfx1250 (MI455X): all GEMMs via v_wmma_f32_16x16x32_bf16.
// B=4, S=1024, D=1024, H=16, HD=64, FF=4096. Compute-bound (~200 GFLOP vs ~5us of HBM
// traffic at 23.3 TB/s), so everything is routed through bf16 WMMA with f32 accumulate.
// GEMM uses 128x64x32 tiles (wave = 4 M-subtiles) to cut weight-panel re-reads and
// amortize B-fragment LDS reads. (Probe note: __builtin_amdgcn_global_load_async_to_lds_b128
// exists here but takes an int4 descriptor as arg0, not pointers — staging stays on the
// global_load_b128 + ds_store_b128 path, which is already wide.)

#define B_   4
#define S_   1024
#define D_   1024
#define H_   16
#define HD_  64
#define FF_  4096
#define EPS_ 1e-5f
#define LMIN_ 0.5f
#define LMAX_ 0.5f

typedef __attribute__((ext_vector_type(16))) __bf16 bf16x16;
typedef __attribute__((ext_vector_type(8)))  float  v8f;

static __device__ __forceinline__ unsigned short f2bf(float f) {
    union { float f; unsigned int u; } c; c.f = f;
    unsigned int u = c.u;
    unsigned int r = u + 0x7fffu + ((u >> 16) & 1u);   // round-to-nearest-even
    return (unsigned short)(r >> 16);
}
static __device__ __forceinline__ __bf16 us2bf(unsigned short u) {
    union { unsigned short u; __bf16 b; } c; c.u = u; return c.b;
}
static __device__ __forceinline__ v8f vzero() {
    v8f z = {0.f,0.f,0.f,0.f,0.f,0.f,0.f,0.f}; return z;
}
static __device__ __forceinline__ v8f wmma_bf16(bf16x16 a, bf16x16 b, v8f c) {
    return __builtin_amdgcn_wmma_f32_16x16x32_bf16(false, a, false, b, (short)0, c, false, false);
}

// 16-byte global -> LDS copy (lowers to global_load_b128 + ds_store_b128).
static __device__ __forceinline__ void cp16(unsigned short* dst, const unsigned short* src) {
    *reinterpret_cast<uint4*>(dst) = *reinterpret_cast<const uint4*>(src);
}

// A fragment (16x32 bf16): lane L holds row M=L&15; K base = (L&16)?8:0;
// elements 0..7 -> K=kb+i, elements 8..15 -> K=kb+16+i.  (ISA 7.12.2)
static __device__ __forceinline__ bf16x16 fragA(const unsigned short* t, int stride,
                                                int r0, int c0, int lane) {
    const int m  = lane & 15;
    const int kb = (lane & 16) ? 8 : 0;
    const unsigned short* p = t + (r0 + m) * stride + c0 + kb;
    bf16x16 a;
#pragma unroll
    for (int i = 0; i < 8; i++) a[i]     = us2bf(p[i]);
#pragma unroll
    for (int i = 0; i < 8; i++) a[8 + i] = us2bf(p[16 + i]);
    return a;
}

// B fragment (32x16 bf16) read from an LDS tile stored [N][K] (transposed), so the 16
// K-elements are contiguous: lane L -> N=L&15, K base=(L&16)?8:0; elems 0..7 K=kb+i, 8..15 K=kb+16+i.
static __device__ __forceinline__ bf16x16 fragB_nk(const unsigned short* t, int stride,
                                                   int n0, int c0, int lane) {
    const int n  = lane & 15;
    const int kb = (lane & 16) ? 8 : 0;
    const unsigned short* p = t + (n0 + n) * stride + c0 + kb;
    bf16x16 b;
#pragma unroll
    for (int i = 0; i < 8; i++) b[i]     = us2bf(p[i]);
#pragma unroll
    for (int i = 0; i < 8; i++) b[8 + i] = us2bf(p[16 + i]);
    return b;
}

// ---- 64x64 bf16 tile loaders (128 threads) -------------------------------------------
static __device__ __forceinline__ void load_tile(unsigned short* dst, const unsigned short* src,
                                                 size_t row0, int ld, int col0, int tid, int nthr) {
    for (int idx = tid; idx < 512; idx += nthr) {      // 512 x 16B = 64x64 bf16
        int r = idx >> 3, c = (idx & 7) * 8;
        cp16(&dst[r * 64 + c], src + (row0 + (size_t)r) * (size_t)ld + col0 + c);
    }
}
static __device__ __forceinline__ void load_tile_T(unsigned short* dst, const unsigned short* src,
                                                   size_t row0, int ld, int col0, int tid, int nthr) {
    for (int idx = tid; idx < 512; idx += nthr) {
        int r = idx >> 3, c = (idx & 7) * 8;
        uint4 v = *reinterpret_cast<const uint4*>(src + (row0 + (size_t)r) * (size_t)ld + col0 + c);
        const unsigned short* pv = reinterpret_cast<const unsigned short*>(&v);
#pragma unroll
        for (int j = 0; j < 8; j++) dst[(c + j) * 64 + r] = pv[j];
    }
}

// ======================================================================================
// LayerNorm -> bf16 (+ optional tanh bf16)
// ======================================================================================
__global__ void layernorm_bf16_kernel(const float* __restrict__ x, const float* __restrict__ g,
                                      const float* __restrict__ b,
                                      unsigned short* __restrict__ obf,
                                      unsigned short* __restrict__ otanh, int C) {
    __shared__ float s1[256], s2[256];
    const int row = blockIdx.x, tid = threadIdx.x;
    const size_t base = (size_t)row * C;
    float sum = 0.f, sq = 0.f;
    for (int i = tid; i < C; i += 256) { float v = x[base + i]; sum += v; sq += v * v; }
    s1[tid] = sum; s2[tid] = sq; __syncthreads();
    for (int st = 128; st > 0; st >>= 1) {
        if (tid < st) { s1[tid] += s1[tid + st]; s2[tid] += s2[tid + st]; }
        __syncthreads();
    }
    const float mu = s1[0] / C;
    const float var = s2[0] / C - mu * mu;
    const float rstd = rsqrtf(var + EPS_);
    for (int i = tid; i < C; i += 256) {
        float y = (x[base + i] - mu) * rstd * g[i] + b[i];
        obf[base + i] = f2bf(y);
        if (otanh) otanh[base + i] = f2bf(tanhf(y));
    }
}

// ======================================================================================
// LayerNorm + exact GELU, with optional residual-add f32 output or bf16 (+tanh) outputs
// ======================================================================================
__global__ void ln_gelu_kernel(const float* __restrict__ hin, const float* __restrict__ g,
                               const float* __restrict__ b, const float* __restrict__ resid,
                               float* __restrict__ outF,
                               unsigned short* __restrict__ obf,
                               unsigned short* __restrict__ otanh, int C) {
    __shared__ float s1[256], s2[256];
    const int row = blockIdx.x, tid = threadIdx.x;
    const size_t base = (size_t)row * C;
    float sum = 0.f, sq = 0.f;
    for (int i = tid; i < C; i += 256) { float v = hin[base + i]; sum += v; sq += v * v; }
    s1[tid] = sum; s2[tid] = sq; __syncthreads();
    for (int st = 128; st > 0; st >>= 1) {
        if (tid < st) { s1[tid] += s1[tid + st]; s2[tid] += s2[tid + st]; }
        __syncthreads();
    }
    const float mu = s1[0] / C;
    const float var = s2[0] / C - mu * mu;
    const float rstd = rsqrtf(var + EPS_);
    for (int i = tid; i < C; i += 256) {
        float t  = (hin[base + i] - mu) * rstd * g[i] + b[i];
        float ge = 0.5f * t * (1.f + erff(t * 0.70710678118f));
        if (outF)  outF[base + i]  = resid[base + i] + ge;
        if (obf)   obf[base + i]   = f2bf(ge);
        if (otanh) otanh[base + i] = f2bf(tanhf(ge));
    }
}

// ======================================================================================
// Generic WMMA GEMM: C = A(bf16)[M,K] @ W(f32->bf16)[K,N] (+ fused A2@J second chain)
// modes: 1 = write bf16(acc+bias) and optional bf16(tanh(.)) ........ (Q/K/V proj)
//        2 = outF = h~ + lam_eff * (A2@J) * tanh(h~), h~ = acc+bias .. (QBNN FFN)
//        3 = outF = resid + acc + bias ............................... (wo + residual)
// Tiles 128x64x32, 256 threads = 8 waves, each wave: 4 M-subtiles x 1 N-subtile
// (B fragment reused 4x per LDS read; weight panel read M/128 times instead of M/64).
// ======================================================================================
__global__ __launch_bounds__(256) void gemm_qbnn_kernel(
    const unsigned short* __restrict__ A,  const unsigned short* __restrict__ A2,
    const float* __restrict__ W,           const float* __restrict__ J,
    const float* __restrict__ bias,        const float* __restrict__ lam_ptr,
    const float* __restrict__ resid,
    float* __restrict__ outF, unsigned short* __restrict__ outBF,
    unsigned short* __restrict__ outT,
    int M, int N, int K, int lda, int ldb, int ldc, int mode)
{
    __shared__ __align__(16) unsigned short lA [128 * 32];
    __shared__ __align__(16) unsigned short lA2[128 * 32];
    __shared__ __align__(16) unsigned short lB [64 * 32];   // stored [n][k]
    __shared__ __align__(16) unsigned short lJ [64 * 32];   // stored [n][k]

    const int tid = threadIdx.x, lane = tid & 31, w = tid >> 5;
    const int nsub = w & 3, mgrp = w >> 2;                  // wave: rows mgrp*64 .. +64
    const int N0 = blockIdx.x * 64, M0 = blockIdx.y * 128;
    const bool fused = (mode == 2);

    float lam_eff = 0.f;
    if (fused) {
        const float lb = lam_ptr[0];
        const float ln = 1.f / (1.f + __expf(-lb));
        lam_eff = LMIN_ + (LMAX_ - LMIN_) * (ln * 0.7f + 0.5f * 0.3f);  // dyn factor 0.5
    }

    v8f acc1[4], acc2[4];
#pragma unroll
    for (int i = 0; i < 4; i++) { acc1[i] = vzero(); acc2[i] = vzero(); }

    for (int k0 = 0; k0 < K; k0 += 32) {
        __syncthreads();
        {   // A (and A2) tile 128x32 bf16: 512 x 16B, two per thread
#pragma unroll
            for (int t = 0; t < 2; t++) {
                const int idx = tid + t * 256;
                const int r = idx >> 2, c = (idx & 3) * 8;
                cp16(&lA[r * 32 + c], A + (size_t)(M0 + r) * lda + k0 + c);
                if (fused)
                    cp16(&lA2[r * 32 + c], A2 + (size_t)(M0 + r) * lda + k0 + c);
            }
        }
        {   // W tile (32x64 f32 -> bf16, transposed into [n][k]): 512 float4
#pragma unroll
            for (int t = 0; t < 2; t++) {
                const int fidx = tid + t * 256;
                const int kk = fidx >> 4, nb = (fidx & 15) * 4;
                float4 wv = *reinterpret_cast<const float4*>(W + (size_t)(k0 + kk) * ldb + N0 + nb);
                lB[(nb + 0) * 32 + kk] = f2bf(wv.x);
                lB[(nb + 1) * 32 + kk] = f2bf(wv.y);
                lB[(nb + 2) * 32 + kk] = f2bf(wv.z);
                lB[(nb + 3) * 32 + kk] = f2bf(wv.w);
                if (fused) {
                    float4 jv = *reinterpret_cast<const float4*>(J + (size_t)(k0 + kk) * ldb + N0 + nb);
                    lJ[(nb + 0) * 32 + kk] = f2bf(jv.x);
                    lJ[(nb + 1) * 32 + kk] = f2bf(jv.y);
                    lJ[(nb + 2) * 32 + kk] = f2bf(jv.z);
                    lJ[(nb + 3) * 32 + kk] = f2bf(jv.w);
                }
            }
        }
        if (k0 + 32 < K)  // pull next weight K-tile into cache (global_prefetch_b8)
            __builtin_prefetch(W + (size_t)(k0 + 32) * ldb + N0, 0, 1);
        __syncthreads();

        const bf16x16 bfr = fragB_nk(lB, 32, nsub * 16, 0, lane);
#pragma unroll
        for (int mi = 0; mi < 4; mi++) {
            bf16x16 af = fragA(lA, 32, (mgrp * 4 + mi) * 16, 0, lane);
            acc1[mi] = wmma_bf16(af, bfr, acc1[mi]);
        }
        if (fused) {
            const bf16x16 jfr = fragB_nk(lJ, 32, nsub * 16, 0, lane);
#pragma unroll
            for (int mi = 0; mi < 4; mi++) {
                bf16x16 a2f = fragA(lA2, 32, (mgrp * 4 + mi) * 16, 0, lane);
                acc2[mi] = wmma_bf16(a2f, jfr, acc2[mi]);
            }
        }
    }

    // epilogue (C fragment: M = v + (lane&16 ? 8:0), N = lane&15)
    const int col = N0 + nsub * 16 + (lane & 15);
    const float bb = bias ? bias[col] : 0.f;
    const int mo = (lane & 16) ? 8 : 0;
#pragma unroll
    for (int mi = 0; mi < 4; mi++) {
        const int rbase = M0 + (mgrp * 4 + mi) * 16 + mo;
#pragma unroll
        for (int v = 0; v < 8; v++) {
            const size_t idx = (size_t)(rbase + v) * ldc + col;
            const float a1 = acc1[mi][v] + bb;
            if (mode == 1) {
                outBF[idx] = f2bf(a1);
                if (outT) outT[idx] = f2bf(tanhf(a1));
            } else if (mode == 2) {
                outF[idx] = a1 + lam_eff * acc2[mi][v] * tanhf(a1);
            } else {
                outF[idx] = resid[idx] + a1;
            }
        }
    }
}

// ======================================================================================
// Flash attention with QBNN delta term. One block per (b, h, 64-query tile); 4 waves.
// scores = Q.K^T/sqrt(HD) + lam * (tanh(Q)@J_h) . tanh(K)^T ; online softmax ; O += P.V
// ======================================================================================
__global__ __launch_bounds__(128) void attn_kernel(
    const unsigned short* __restrict__ Q,  const unsigned short* __restrict__ tQ,
    const unsigned short* __restrict__ Kb, const unsigned short* __restrict__ tK,
    const unsigned short* __restrict__ V,  const float* __restrict__ Jh,
    const float* __restrict__ lam_ptr,     unsigned short* __restrict__ O)
{
    __shared__ __align__(16) unsigned short lQ  [64 * 64];
    __shared__ __align__(16) unsigned short lTQJ[64 * 64];
    __shared__ __align__(16) unsigned short lK  [64 * 64];   // also tQ in phase 1
    __shared__ __align__(16) unsigned short lTK [64 * 64];   // also J^T in phase 1
    __shared__ __align__(16) unsigned short lVt [64 * 64];   // V transposed [d][key]
    __shared__ __align__(16) unsigned short lP  [4 * 16 * 64];

    const int tid = threadIdx.x, lane = tid & 31, w = tid >> 5;
    const int q0 = blockIdx.x * 64;
    const int bh = blockIdx.y;
    const int bb = bh / H_, hh = bh % H_;
    const float lam = lam_ptr[0];
    const float inv_scale = 0.125f;                 // 1/sqrt(64)
    const size_t rowOff = (size_t)bb * S_;
    const int colOff = hh * HD_;
    const int nn = lane & 15, mo = (lane & 16) ? 8 : 0;

    // ---- phase 1: tQJ tile = tanh(Q) @ J_h -------------------------------------------
    load_tile(lQ, Q,  rowOff + q0, D_, colOff, tid, 128);
    load_tile(lK, tQ, rowOff + q0, D_, colOff, tid, 128);    // lK temporarily = tanh(Q)
    for (int idx = tid; idx < 64 * 64; idx += 128) {         // J_h -> bf16, transposed [n][k]
        const int k = idx >> 6, n = idx & 63;
        lTK[n * 64 + k] = f2bf(Jh[(size_t)hh * HD_ * HD_ + idx]);
    }
    __syncthreads();
    {
        v8f t[4]; t[0] = vzero(); t[1] = vzero(); t[2] = vzero(); t[3] = vzero();
#pragma unroll
        for (int ns = 0; ns < 4; ns++)
#pragma unroll
            for (int kk = 0; kk < 64; kk += 32) {
                bf16x16 a  = fragA(lK, 64, w * 16, kk, lane);
                bf16x16 bm = fragB_nk(lTK, 64, ns * 16, kk, lane);
                t[ns] = wmma_bf16(a, bm, t[ns]);
            }
#pragma unroll
        for (int ns = 0; ns < 4; ns++)
#pragma unroll
            for (int v = 0; v < 8; v++)
                lTQJ[(w * 16 + mo + v) * 64 + ns * 16 + nn] = f2bf(t[ns][v]);
    }

    float rm[8], rl[8];
#pragma unroll
    for (int v = 0; v < 8; v++) { rm[v] = -1e30f; rl[v] = 0.f; }
    v8f o[4]; o[0] = vzero(); o[1] = vzero(); o[2] = vzero(); o[3] = vzero();

    // ---- phase 2: stream key tiles ---------------------------------------------------
    for (int kt = 0; kt < S_ / 64; kt++) {
        __syncthreads();
        load_tile  (lK,  Kb, rowOff + kt * 64, D_, colOff, tid, 128);
        load_tile  (lTK, tK, rowOff + kt * 64, D_, colOff, tid, 128);
        load_tile_T(lVt, V,  rowOff + kt * 64, D_, colOff, tid, 128);
        __syncthreads();

        v8f sa[4], da[4];
#pragma unroll
        for (int i = 0; i < 4; i++) { sa[i] = vzero(); da[i] = vzero(); }
#pragma unroll
        for (int ns = 0; ns < 4; ns++)
#pragma unroll
            for (int kk = 0; kk < 64; kk += 32) {
                sa[ns] = wmma_bf16(fragA(lQ,   64, w * 16, kk, lane),
                                   fragB_nk(lK,  64, ns * 16, kk, lane), sa[ns]);
                da[ns] = wmma_bf16(fragA(lTQJ, 64, w * 16, kk, lane),
                                   fragB_nk(lTK, 64, ns * 16, kk, lane), da[ns]);
            }

        float tmax[8];
#pragma unroll
        for (int v = 0; v < 8; v++) tmax[v] = -1e30f;
#pragma unroll
        for (int ns = 0; ns < 4; ns++)
#pragma unroll
            for (int v = 0; v < 8; v++) {
                float s = sa[ns][v] * inv_scale + lam * da[ns][v];
                sa[ns][v] = s;
                tmax[v] = fmaxf(tmax[v], s);
            }
#pragma unroll
        for (int v = 0; v < 8; v++)
#pragma unroll
            for (int off = 1; off < 16; off <<= 1)
                tmax[v] = fmaxf(tmax[v], __shfl_xor(tmax[v], off, 16));

        float alpha[8], rsum[8];
#pragma unroll
        for (int v = 0; v < 8; v++) {
            float nm = fmaxf(rm[v], tmax[v]);
            alpha[v] = __expf(rm[v] - nm);
            rm[v] = nm; rsum[v] = 0.f;
        }
#pragma unroll
        for (int ns = 0; ns < 4; ns++)
#pragma unroll
            for (int v = 0; v < 8; v++) {
                float p = __expf(sa[ns][v] - rm[v]);
                sa[ns][v] = p; rsum[v] += p;
            }
#pragma unroll
        for (int v = 0; v < 8; v++)
#pragma unroll
            for (int off = 1; off < 16; off <<= 1)
                rsum[v] += __shfl_xor(rsum[v], off, 16);
#pragma unroll
        for (int v = 0; v < 8; v++) rl[v] = rl[v] * alpha[v] + rsum[v];
#pragma unroll
        for (int i = 0; i < 4; i++)
#pragma unroll
            for (int v = 0; v < 8; v++) o[i][v] *= alpha[v];

        // P (C-layout) -> LDS -> A-layout fragments
#pragma unroll
        for (int ns = 0; ns < 4; ns++)
#pragma unroll
            for (int v = 0; v < 8; v++)
                lP[w * 1024 + (mo + v) * 64 + ns * 16 + nn] = f2bf(sa[ns][v]);
        __syncthreads();

#pragma unroll
        for (int ds = 0; ds < 4; ds++)
#pragma unroll
            for (int kk = 0; kk < 64; kk += 32)
                o[ds] = wmma_bf16(fragA(lP + w * 1024, 64, 0, kk, lane),
                                  fragB_nk(lVt, 64, ds * 16, kk, lane), o[ds]);
    }

    // ---- epilogue: normalize, store bf16 in (b, s, h*HD + d) layout ------------------
#pragma unroll
    for (int ds = 0; ds < 4; ds++)
#pragma unroll
        for (int v = 0; v < 8; v++) {
            const float val = o[ds][v] / rl[v];
            const size_t row = rowOff + q0 + w * 16 + mo + v;
            O[row * D_ + colOff + ds * 16 + nn] = f2bf(val);
        }
}

// ======================================================================================
// Host-side orchestration
// ======================================================================================
extern "C" void kernel_launch(void* const* d_in, const int* in_sizes, int n_in,
                              void* d_out, int out_size, void* d_ws, size_t ws_size,
                              hipStream_t stream) {
    (void)in_sizes; (void)n_in; (void)out_size; (void)ws_size;
    const float* x       = (const float*)d_in[0];
    const float* wq      = (const float*)d_in[1];  const float* bq   = (const float*)d_in[2];
    const float* wk      = (const float*)d_in[3];  const float* bk   = (const float*)d_in[4];
    const float* wv      = (const float*)d_in[5];  const float* bv   = (const float*)d_in[6];
    const float* wo      = (const float*)d_in[7];  const float* bo   = (const float*)d_in[8];
    const float* J_attn  = (const float*)d_in[9];  const float* lamA = (const float*)d_in[10];
    const float* g_attn  = (const float*)d_in[11]; const float* b_attn = (const float*)d_in[12];
    const float* W1      = (const float*)d_in[13]; const float* b1   = (const float*)d_in[14];
    const float* J1      = (const float*)d_in[15]; const float* lam1 = (const float*)d_in[16];
    const float* g1      = (const float*)d_in[17]; const float* be1  = (const float*)d_in[18];
    const float* W2      = (const float*)d_in[19]; const float* b2   = (const float*)d_in[20];
    const float* J2      = (const float*)d_in[21]; const float* lam2 = (const float*)d_in[22];
    const float* g2      = (const float*)d_in[23]; const float* be2  = (const float*)d_in[24];
    const float* g_ffn   = (const float*)d_in[25]; const float* b_ffn = (const float*)d_in[26];
    float* out = (float*)d_out;

    const size_t ND = (size_t)B_ * S_ * D_;    // 4.19M
    const size_t NF = (size_t)B_ * S_ * FF_;   // 16.8M
    char* ws = (char*)d_ws; size_t off = 0;
    auto carve = [&](size_t bytes) -> void* {
        void* p = ws + off;
        off += (bytes + 255) & ~(size_t)255;
        return p;
    };
    unsigned short* xn  = (unsigned short*)carve(ND * 2);
    unsigned short* Qb  = (unsigned short*)carve(ND * 2);
    unsigned short* tQb = (unsigned short*)carve(ND * 2);
    unsigned short* Kb  = (unsigned short*)carve(ND * 2);
    unsigned short* tKb = (unsigned short*)carve(ND * 2);
    unsigned short* Vb  = (unsigned short*)carve(ND * 2);
    unsigned short* Ab  = (unsigned short*)carve(ND * 2);
    float*          x2  = (float*)carve(ND * 4);
    unsigned short* hn  = (unsigned short*)carve(ND * 2);
    unsigned short* thn = (unsigned short*)carve(ND * 2);
    float*          hh1 = (float*)carve(NF * 4);
    unsigned short* h1  = (unsigned short*)carve(NF * 2);
    unsigned short* th1 = (unsigned short*)carve(NF * 2);
    float*          hh2 = (float*)carve(ND * 4);

    const int rows = B_ * S_;                       // 4096
    const dim3 blk(256);
    const dim3 gDD(D_ / 64, rows / 128);            // N=1024 tiles x M/128 tiles
    const dim3 gDF(FF_ / 64, rows / 128);           // N=4096 tiles x M/128 tiles

    // attention: xn = LN(x); Q/K/V (+tanh) projections; flash attention; x2 = x + attn@wo
    layernorm_bf16_kernel<<<rows, 256, 0, stream>>>(x, g_attn, b_attn, xn, nullptr, D_);
    gemm_qbnn_kernel<<<gDD, blk, 0, stream>>>(xn, nullptr, wq, nullptr, bq, nullptr, nullptr,
                                              nullptr, Qb, tQb, rows, D_, D_, D_, D_, D_, 1);
    gemm_qbnn_kernel<<<gDD, blk, 0, stream>>>(xn, nullptr, wk, nullptr, bk, nullptr, nullptr,
                                              nullptr, Kb, tKb, rows, D_, D_, D_, D_, D_, 1);
    gemm_qbnn_kernel<<<gDD, blk, 0, stream>>>(xn, nullptr, wv, nullptr, bv, nullptr, nullptr,
                                              nullptr, Vb, nullptr, rows, D_, D_, D_, D_, D_, 1);
    attn_kernel<<<dim3(S_ / 64, B_ * H_), 128, 0, stream>>>(Qb, tQb, Kb, tKb, Vb,
                                                            J_attn, lamA, Ab);
    gemm_qbnn_kernel<<<gDD, blk, 0, stream>>>(Ab, nullptr, wo, nullptr, bo, nullptr, x,
                                              x2, nullptr, nullptr, rows, D_, D_, D_, D_, D_, 3);

    // QBNN FFN: hn = LN(x2) (+tanh); fused h@W1 / tanh(h)@J1; LN+GELU; layer 2; final
    layernorm_bf16_kernel<<<rows, 256, 0, stream>>>(x2, g_ffn, b_ffn, hn, thn, D_);
    gemm_qbnn_kernel<<<gDF, blk, 0, stream>>>(hn, thn, W1, J1, b1, lam1, nullptr,
                                              hh1, nullptr, nullptr, rows, FF_, D_, D_, FF_, FF_, 2);
    ln_gelu_kernel<<<rows, 256, 0, stream>>>(hh1, g1, be1, nullptr, nullptr, h1, th1, FF_);
    gemm_qbnn_kernel<<<gDD, blk, 0, stream>>>(h1, th1, W2, J2, b2, lam2, nullptr,
                                              hh2, nullptr, nullptr, rows, D_, FF_, FF_, D_, D_, 2);
    ln_gelu_kernel<<<rows, 256, 0, stream>>>(hh2, g2, be2, x2, out, nullptr, nullptr, D_);
}
